// PixelNeRFNet_5712306504494
// MI455X (gfx1250) — matrix operand order repaired and checked by reading, not compiled, not run on hardware
//
#include <hip/hip_runtime.h>
#include <hip/hip_bf16.h>

typedef __attribute__((ext_vector_type(16))) _Float16 v16h;
typedef __attribute__((ext_vector_type(8)))  _Float16 v8h;
typedef __attribute__((ext_vector_type(8)))  float    v8f;
typedef __attribute__((ext_vector_type(4)))  unsigned int ui32x4;
typedef __attribute__((ext_vector_type(8)))  int      i32x8;
typedef __attribute__((ext_vector_type(4)))  int      i32x4;

// Problem constants (from reference setup_inputs)
#define S_    2
#define VIN_  3
#define VT_   4
#define C_    16
#define D_    64
#define H_    96
#define W_    96
#define h_    48
#define w_    48
#define SPR_  64
#define DHW_  (D_*H_*W_)                      // 589824
#define NPTS_ ((long)S_*VT_*h_*w_*SPR_)       // 1179648
#define NRAYS_ (S_*VT_*h_*w_)                 // 18432

// ---------------------------------------------------------------------------
// K1: ray generation + projection into each input view + trilinear sampling of
// the (C,D,H,W) feature volume, mean over Vin. One thread per sample point.
// ---------------------------------------------------------------------------
__global__ __launch_bounds__(256) void sample_kernel(
    const float* __restrict__ enc,         // (S,Vin,C,D,H,W)
    const float* __restrict__ input_poses, // (S,Vin,4,4)
    const float* __restrict__ target_poses,// (S,Vt,4,4)
    const float* __restrict__ focal_p,
    const float* __restrict__ zn_p,
    const float* __restrict__ zf_p,
    _Float16* __restrict__ feats)          // (NPTS, 16) f16
{
    long p = (long)blockIdx.x * 256 + threadIdx.x;
    if (p >= NPTS_) return;

    int d   = (int)(p & 63);
    long r  = p >> 6;
    int x   = (int)(r % w_);  r /= w_;
    int y   = (int)(r % h_);  r /= h_;
    int vt  = (int)(r & 3);
    int s   = (int)(r >> 2);

    float focal = *focal_p;
    float zn = *zn_p, zf = *zf_p;

    // camera ray (target view)
    float ys = ((float)y + 0.5f) / h_ * 2.0f - 1.0f;
    float xs = ((float)x + 0.5f) / w_ * 2.0f - 1.0f;
    float dcx = xs / focal, dcy = -ys / focal, dcz = -1.0f;

    const float* tp = target_poses + ((s * VT_ + vt) * 16);
    float dx0 = tp[0]*dcx + tp[1]*dcy + tp[2]*dcz;
    float dy0 = tp[4]*dcx + tp[5]*dcy + tp[6]*dcz;
    float dz0 = tp[8]*dcx + tp[9]*dcy + tp[10]*dcz;
    float ox = tp[3], oy = tp[7], oz = tp[11];

    float t = zn + (zf - zn) * ((float)d * (1.0f / (SPR_ - 1)));
    float px = ox + t * dx0;
    float py = oy + t * dy0;
    float pz = oz + t * dz0;

    float minz = focal * zn, maxz = focal * zf;
    float inv_zr = 2.0f / (maxz - minz);

    float acc[C_];
#pragma unroll
    for (int c = 0; c < C_; ++c) acc[c] = 0.0f;

    for (int v = 0; v < VIN_; ++v) {
        const float* ip = input_poses + ((s * VIN_ + v) * 16);
        // local = R^T (pt - tr)
        float qx = px - ip[3], qy = py - ip[7], qz = pz - ip[11];
        float lx = ip[0]*qx + ip[4]*qy + ip[8]*qz;
        float ly = ip[1]*qx + ip[5]*qy + ip[9]*qz;
        float lz = ip[2]*qx + ip[6]*qy + ip[10]*qz;
        float z  = -lz;
        float u  =  lx / z * focal;
        float vv =  ly / z * focal;
        float wd = (z - minz) * inv_zr - 1.0f;
        // uvw = (u, -v, -wd); grid x->W, y->H, z->D
        float gx = u, gy = -vv, gz = -wd;
        float ix = ((gx + 1.0f) * W_ - 1.0f) * 0.5f;
        float iy = ((gy + 1.0f) * H_ - 1.0f) * 0.5f;
        float iz = ((gz + 1.0f) * D_ - 1.0f) * 0.5f;
        float x0f = floorf(ix), y0f = floorf(iy), z0f = floorf(iz);

        const float* vol = enc + (size_t)((s * VIN_ + v) * C_) * DHW_;

#pragma unroll
        for (int dz = 0; dz < 2; ++dz)
#pragma unroll
        for (int dy = 0; dy < 2; ++dy)
#pragma unroll
        for (int dxc = 0; dxc < 2; ++dxc) {
            float xc = x0f + dxc, yc = y0f + dy, zc = z0f + dz;
            float wgt = (1.0f - fabsf(ix - xc)) * (1.0f - fabsf(iy - yc)) * (1.0f - fabsf(iz - zc));
            bool valid = (xc >= 0.0f) && (xc < (float)W_) &&
                         (yc >= 0.0f) && (yc < (float)H_) &&
                         (zc >= 0.0f) && (zc < (float)D_);
            if (valid) {
                int xi = (int)xc, yi = (int)yc, zi = (int)zc;
                size_t idx = (size_t)(zi * H_ + yi) * W_ + xi;
                const float* cp = vol + idx;
#pragma unroll
                for (int c = 0; c < C_; ++c)
                    acc[c] += wgt * cp[(size_t)c * DHW_];
            }
        }
    }

    _Float16* fo = feats + p * C_;
#pragma unroll
    for (int c = 0; c < C_; ++c)
        fo[c] = (_Float16)(acc[c] * (1.0f / VIN_));
}

// ---------------------------------------------------------------------------
// TDM helper: 1-row 2-D tensor descriptor moving `nelems` f32 from global to
// an LDS byte offset. clang-23 6-arg builtin form:
//   (uint32x4 g0, int32x8 g1, int32x4 g2, int32x4 g3, int32x8 extra, i32 cpol)
// D# group0: [1:0]=count=1, [63:32]=lds_addr, [120:64]=global_addr, [127:126]=type=2
// D# group1: [17:16]=data_size(2 => 4B), [79:48]=tensor_dim0, [111:80]=tensor_dim1,
//            [127:112]=tile_dim0, [143:128]=tile_dim1, [207:160]=dim0_stride,
//            [255:208]=dim1_stride
// ---------------------------------------------------------------------------
__device__ __forceinline__ void tdm_load_f32_1d(const float* gptr,
                                                unsigned lds_byte_off,
                                                unsigned nelems)
{
    unsigned long long ga = (unsigned long long)(uintptr_t)gptr;
    ui32x4 g0;
    g0[0] = 1u;                                  // count=1, user mode
    g0[1] = lds_byte_off;                        // lds_addr
    g0[2] = (unsigned)(ga & 0xffffffffu);        // global_addr[31:0]
    g0[3] = (unsigned)((ga >> 32) & 0x01ffffffu) // global_addr[56:32]
            | (2u << 30);                        // type = 2 ("image")
    i32x8 g1;
    g1[0] = (int)(2u << 16);                     // data_size = 4 bytes
    g1[1] = (int)((nelems & 0xffffu) << 16);     // tensor_dim0[15:0]
    g1[2] = (int)((nelems >> 16) | (1u << 16));  // tensor_dim0[31:16] | tensor_dim1=1
    g1[3] = (int)((nelems & 0xffffu) << 16);     // tile_dim0 (<=65535)
    g1[4] = 1;                                   // tile_dim1 = 1, tile_dim2 = 0
    g1[5] = (int)nelems;                         // tensor_dim0_stride[31:0]
    g1[6] = (int)((nelems & 0xffffu) << 16);     // stride0[47:32]=0 | stride1[15:0]
    g1[7] = 0;                                   // stride1[47:16]
    i32x4 z4; z4[0] = 0; z4[1] = 0; z4[2] = 0; z4[3] = 0;
    i32x8 z8;
#pragma unroll
    for (int j = 0; j < 8; ++j) z8[j] = 0;
    __builtin_amdgcn_tensor_load_to_lds(g0, g1, z4, z4, z8, 0);
}

__device__ __forceinline__ unsigned lds_offset_of(const void* shared_ptr)
{
    // Flat-aperture rule: the LDS offset is the low 32 bits of the shared address.
    return (unsigned)(uintptr_t)shared_ptr;
}

// ---------------------------------------------------------------------------
// K2: WMMA MLP 16->64->64->17.  Block = 256 threads = 8 waves; each wave
// processes one 16-point tile with v_wmma_f32_16x16x32_f16.
// Raw f32 weights DMA'd into LDS by the Tensor Data Mover, then converted to
// transposed f16 fragment layouts (contiguous per-lane B-fragments).
// ---------------------------------------------------------------------------
__global__ __launch_bounds__(256) void mlp_kernel(
    const _Float16* __restrict__ feats,
    const float* __restrict__ W1, const float* __restrict__ b1,
    const float* __restrict__ W2, const float* __restrict__ b2,
    const float* __restrict__ W3, const float* __restrict__ b3,
    float* __restrict__ density, _Float16* __restrict__ rgbd)
{
    __shared__ float    wraw[1024 + 4096 + 1088]; // raw W1|W2|W3 via TDM
    __shared__ _Float16 w1p[64 * 32];   // [n][k] of W1 (16x64), K padded to 32 w/ zeros
    __shared__ _Float16 w2t[64 * 64];   // [n][k] of W2 (64x64)
    __shared__ _Float16 w3t[32 * 64];   // [n][k] of W3 (64x17), n padded to 32
    __shared__ float bs1[64], bs2[64], bs3[32];
    __shared__ _Float16 xbuf[8][16 * 64]; // per-wave inter-layer staging

    const int tid = threadIdx.x;

    // ---- TDM: wave-level DMA of the three weight matrices into LDS --------
    if (tid == 0) {
        tdm_load_f32_1d(W1, lds_offset_of(&wraw[0]),    1024);
        tdm_load_f32_1d(W2, lds_offset_of(&wraw[1024]), 4096);
        tdm_load_f32_1d(W3, lds_offset_of(&wraw[5120]), 1088);
        __builtin_amdgcn_s_wait_tensorcnt(0);
    }
    if (tid < 64) { bs1[tid] = b1[tid]; bs2[tid] = b2[tid]; }
    if (tid < 32) { bs3[tid] = (tid < 17) ? b3[tid] : 0.0f; }
    __syncthreads();

    // ---- convert raw f32 -> transposed f16 fragment layouts ---------------
    const float* W1r = &wraw[0];
    const float* W2r = &wraw[1024];
    const float* W3r = &wraw[5120];
    for (int i = tid; i < 64 * 32; i += 256) {      // w1p[n][k], zero for k>=16
        int n = i >> 5, k = i & 31;
        w1p[i] = (k < 16) ? (_Float16)W1r[k * 64 + n] : (_Float16)0.0f;
    }
    for (int i = tid; i < 64 * 64; i += 256) {      // w2t[n][k]
        int n = i >> 6, k = i & 63;
        w2t[i] = (_Float16)W2r[k * 64 + n];
    }
    for (int i = tid; i < 32 * 64; i += 256) {      // w3t[n][k], zero rows n>=17
        int n = i >> 6, k = i & 63;
        w3t[i] = (n < 17) ? (_Float16)W3r[k * 17 + n] : (_Float16)0.0f;
    }
    __syncthreads();

    const int wave = tid >> 5;
    const int lane = tid & 31;
    const int mn   = lane & 15;   // row (A) / column (B,C,D) index for this lane
    const int hi   = lane >> 4;   // lane-half selector
    const long tile = (long)blockIdx.x * 8 + wave;
    _Float16* xw = &xbuf[wave][0];

    // ---------------- layer 1: feats(16x16, K padded to 32) @ W1 -> 16x64 ---
    v16h a1;
#pragma unroll
    for (int j = 0; j < 16; ++j) a1[j] = (_Float16)0.0f;
    {
        // A layout: lane<16 holds K=0..7 (pad 16..23); lane>=16 K=8..15 (pad 24..31)
        const v8h* fp = (const v8h*)(feats + (tile * 16 + mn) * 16 + hi * 8);
        v8h fv = *fp;
#pragma unroll
        for (int j = 0; j < 8; ++j) a1[j] = fv[j];   // halves 8..15 stay zero
    }

    v8f c1[4];
#pragma unroll
    for (int nt = 0; nt < 4; ++nt) {
        // B layout: lane<16 holds K=0..7 (pad 16..23); lane>=16 K=8..15 (pad 24..31)
        const _Float16* wp = &w1p[(nt * 16 + mn) * 32 + hi * 8];
        v8h blo = *(const v8h*)(wp + 0);
        v8h bhi = *(const v8h*)(wp + 16);
        v16h b;
#pragma unroll
        for (int j = 0; j < 8; ++j) { b[j] = blo[j]; b[8 + j] = bhi[j]; }
        v8f cz;
#pragma unroll
        for (int j = 0; j < 8; ++j) cz[j] = 0.0f;
        c1[nt] = __builtin_amdgcn_wmma_f32_16x16x32_f16(
            false, a1, false, b, (short)0, cz, false, false);
    }
    // bias + relu + stage x1 (16x64) to LDS in row-major [m][k]
#pragma unroll
    for (int nt = 0; nt < 4; ++nt) {
        float bb = bs1[nt * 16 + mn];
#pragma unroll
        for (int v = 0; v < 8; ++v) {
            float val = fmaxf(c1[nt][v] + bb, 0.0f);
            xw[(v + 8 * hi) * 64 + nt * 16 + mn] = (_Float16)val;
        }
    }

    // ---------------- layer 2: x1(16x64) @ W2(64x64) -> 16x64 --------------
    v16h a2[2];
#pragma unroll
    for (int ks = 0; ks < 2; ++ks) {
        int kb = ks * 32 + hi * 8;
        v8h lo = *(const v8h*)&xw[mn * 64 + kb];
        v8h hp = *(const v8h*)&xw[mn * 64 + kb + 16];
#pragma unroll
        for (int j = 0; j < 8; ++j) { a2[ks][j] = lo[j]; a2[ks][8 + j] = hp[j]; }
    }
    v8f c2[4];
#pragma unroll
    for (int nt = 0; nt < 4; ++nt) {
#pragma unroll
        for (int j = 0; j < 8; ++j) c2[nt][j] = 0.0f;
#pragma unroll
        for (int ks = 0; ks < 2; ++ks) {
            int kofs = ks * 32 + hi * 8;
            const _Float16* wp = &w2t[(nt * 16 + mn) * 64 + kofs];
            v8h blo = *(const v8h*)(wp + 0);
            v8h bhi = *(const v8h*)(wp + 16);
            v16h b;
#pragma unroll
            for (int j = 0; j < 8; ++j) { b[j] = blo[j]; b[8 + j] = bhi[j]; }
            c2[nt] = __builtin_amdgcn_wmma_f32_16x16x32_f16(
                false, a2[ks], false, b, (short)0, c2[nt], false, false);
        }
    }
#pragma unroll
    for (int nt = 0; nt < 4; ++nt) {
        float bb = bs2[nt * 16 + mn];
#pragma unroll
        for (int v = 0; v < 8; ++v) {
            float val = fmaxf(c2[nt][v] + bb, 0.0f);
            xw[(v + 8 * hi) * 64 + nt * 16 + mn] = (_Float16)val;
        }
    }

    // ---------------- layer 3: x2(16x64) @ W3(64x17->32) -> 16x32 ----------
    v16h a3[2];
#pragma unroll
    for (int ks = 0; ks < 2; ++ks) {
        int kb = ks * 32 + hi * 8;
        v8h lo = *(const v8h*)&xw[mn * 64 + kb];
        v8h hp = *(const v8h*)&xw[mn * 64 + kb + 16];
#pragma unroll
        for (int j = 0; j < 8; ++j) { a3[ks][j] = lo[j]; a3[ks][8 + j] = hp[j]; }
    }
    v8f c3[2];
#pragma unroll
    for (int nt = 0; nt < 2; ++nt) {
#pragma unroll
        for (int j = 0; j < 8; ++j) c3[nt][j] = 0.0f;
#pragma unroll
        for (int ks = 0; ks < 2; ++ks) {
            int kofs = ks * 32 + hi * 8;
            const _Float16* wp = &w3t[(nt * 16 + mn) * 64 + kofs];
            v8h blo = *(const v8h*)(wp + 0);
            v8h bhi = *(const v8h*)(wp + 16);
            v16h b;
#pragma unroll
            for (int j = 0; j < 8; ++j) { b[j] = blo[j]; b[8 + j] = bhi[j]; }
            c3[nt] = __builtin_amdgcn_wmma_f32_16x16x32_f16(
                false, a3[ks], false, b, (short)0, c3[nt], false, false);
        }
    }

    // Write outputs: col 0 -> relu density (f32); cols 1..16 -> rgb delta (f16)
#pragma unroll
    for (int nt = 0; nt < 2; ++nt) {
        int col = nt * 16 + mn;
        float bb = bs3[col];
#pragma unroll
        for (int v = 0; v < 8; ++v) {
            float val = c3[nt][v] + bb;
            long pnt = tile * 16 + (v + 8 * hi);
            if (col == 0) {
                density[pnt] = fmaxf(val, 0.0f);
            } else if (col < 17) {
                rgbd[pnt * 16 + (col - 1)] = (_Float16)val;
            }
        }
    }
}

// ---------------------------------------------------------------------------
// K3: volume rendering. One thread per (ray, channel).
// wts[d] = alpha[d]*T[d] = exp(-cum_{<d}) - exp(-cum_{<=d})
// ---------------------------------------------------------------------------
__global__ __launch_bounds__(256) void render_kernel(
    const _Float16* __restrict__ feats,
    const float* __restrict__ density,
    const _Float16* __restrict__ rgbd,
    const float* __restrict__ zn_p, const float* __restrict__ zf_p,
    float* __restrict__ renders) // (S*Vt, C, h, w)
{
    int g = blockIdx.x * 256 + threadIdx.x;
    if (g >= NRAYS_ * C_) return;
    int c = g & 15;
    int r = g >> 4;

    float zn = *zn_p, zf = *zf_p;
    float dt = (zf - zn) * (1.0f / (SPR_ - 1));

    long pb = (long)r * SPR_;
    float cum = 0.0f, accum = 0.0f, Tprev = 1.0f;
    for (int d = 0; d < SPR_; ++d) {
        long p = pb + d;
        float den = density[p];
        float delta = (d < SPR_ - 1) ? dt : 1e10f;
        cum += den * delta;
        float Tcur = __expf(-cum);
        float wgt = Tprev - Tcur;
        Tprev = Tcur;
        float rgb = (float)feats[p * 16 + c] + (float)rgbd[p * 16 + c];
        accum += wgt * rgb;
    }
    int xq = r % w_;
    int t1 = r / w_;
    int y  = t1 % h_;
    int sv = t1 / h_;
    renders[((sv * C_ + c) * h_ + y) * w_ + xq] = accum;
}

// ---------------------------------------------------------------------------
// K4: 2x align-corners bilinear upsample (48x48 -> 96x96) into d_out.
// ---------------------------------------------------------------------------
__global__ __launch_bounds__(256) void upsample_kernel(
    const float* __restrict__ renders, float* __restrict__ out)
{
    long o = (long)blockIdx.x * 256 + threadIdx.x;
    if (o >= NPTS_) return;                 // out has same count: 8*16*96*96
    int X = (int)(o % 96);
    long t = o / 96;
    int Y = (int)(t % 96);
    int u = (int)(t / 96);                  // sv*16 + c

    const float scale = 47.0f / 95.0f;
    float py = Y * scale, px = X * scale;
    int y0 = (int)floorf(py), x0 = (int)floorf(px);
    int y1 = min(y0 + 1, 47), x1 = min(x0 + 1, 47);
    float fy = py - (float)y0, fx = px - (float)x0;

    const float* rp = renders + (long)u * (h_ * w_);
    float v00 = rp[y0 * w_ + x0], v01 = rp[y0 * w_ + x1];
    float v10 = rp[y1 * w_ + x0], v11 = rp[y1 * w_ + x1];
    float top = v00 * (1.0f - fx) + v01 * fx;
    float bot = v10 * (1.0f - fx) + v11 * fx;
    out[o] = top * (1.0f - fy) + bot * fy;
}

// ---------------------------------------------------------------------------
extern "C" void kernel_launch(void* const* d_in, const int* in_sizes, int n_in,
                              void* d_out, int out_size, void* d_ws, size_t ws_size,
                              hipStream_t stream) {
    const float* enc  = (const float*)d_in[0];
    const float* ipos = (const float*)d_in[1];
    const float* tpos = (const float*)d_in[2];
    const float* foc  = (const float*)d_in[3];
    const float* zn   = (const float*)d_in[4];
    const float* zf   = (const float*)d_in[5];
    // d_in[6] = samples_per_ray (== 64, hardcoded)
    const float* W1 = (const float*)d_in[7];
    const float* b1 = (const float*)d_in[8];
    const float* W2 = (const float*)d_in[9];
    const float* b2 = (const float*)d_in[10];
    const float* W3 = (const float*)d_in[11];
    const float* b3 = (const float*)d_in[12];

    // workspace layout
    char* ws = (char*)d_ws;
    size_t feats_bytes = (size_t)NPTS_ * C_ * sizeof(_Float16);   // 37,748,736
    size_t dens_bytes  = (size_t)NPTS_ * sizeof(float);           //  4,718,592
    size_t rgbd_bytes  = (size_t)NPTS_ * C_ * sizeof(_Float16);   // 37,748,736
    _Float16* feats   = (_Float16*)ws;
    float*    density = (float*)(ws + feats_bytes);
    _Float16* rgbd    = (_Float16*)(ws + feats_bytes + dens_bytes);
    float*    renders = (float*)(ws + feats_bytes + dens_bytes + rgbd_bytes);

    // K1: 1,179,648 points
    sample_kernel<<<(int)(NPTS_ / 256), 256, 0, stream>>>(
        enc, ipos, tpos, foc, zn, zf, feats);

    // K2: 73,728 tiles of 16 points, 8 waves/block
    mlp_kernel<<<(int)(NPTS_ / 16 / 8), 256, 0, stream>>>(
        feats, W1, b1, W2, b2, W3, b3, density, rgbd);

    // K3: 18,432 rays x 16 channels
    render_kernel<<<(NRAYS_ * C_ + 255) / 256, 256, 0, stream>>>(
        feats, density, rgbd, zn, zf, renders);

    // K4: upsample into d_out
    upsample_kernel<<<(int)(NPTS_ / 256), 256, 0, stream>>>(
        renders, (float*)d_out);
}
